// Phi3Retention_65721589563823
// MI455X (gfx1250) — compile-verified, matrix operand order checked by reading, not run on hardware
//
#include <hip/hip_runtime.h>

typedef __attribute__((ext_vector_type(16))) _Float16 v16h;
typedef __attribute__((ext_vector_type(8)))  _Float16 v8h;
typedef __attribute__((ext_vector_type(4)))  _Float16 v4h;
typedef __attribute__((ext_vector_type(2)))  _Float16 v2h;
typedef __attribute__((ext_vector_type(8)))  float    v8f;
typedef __attribute__((ext_vector_type(4)))  float    v4f;
typedef __attribute__((ext_vector_type(4)))  int      i4;
typedef __attribute__((ext_vector_type(8)))  int      i8;
typedef __attribute__((ext_vector_type(4)))  unsigned u4;

#define Bc   2
#define Lc   1024
#define Dc   3072
#define Hc   32
#define HDc  96
#define Mrows (Bc*Lc)            // 2048
#define SCALING_F 1.1902380714238083f   // sqrt(1 + log(32)/log(4096))
#define QSCALE_F  0.10206207261596575f  // 96^-0.5

// ---------------------------------------------------------------------------
// gfx1250 async global->LDS copy (ASYNCcnt path), with sync fallback.
// Builtin signature (from hipcc diagnostic): (global int4*, lds int4*, Ii, Ii)
// ---------------------------------------------------------------------------
#if __has_builtin(__builtin_amdgcn_global_load_async_to_lds_b128) && \
    __has_builtin(__builtin_amdgcn_s_wait_asynccnt)
#define HAVE_ASYNC_LDS 1
typedef __attribute__((address_space(1))) i4 gi4;
typedef __attribute__((address_space(3))) i4 li4;
__device__ __forceinline__ void async_cp16(void* lds, const void* g) {
  __builtin_amdgcn_global_load_async_to_lds_b128((gi4*)g, (li4*)lds, 0, 0);
}
__device__ __forceinline__ void async_wait0() {
  __builtin_amdgcn_s_wait_asynccnt(0);
}
#else
__device__ __forceinline__ void async_cp16(void* lds, const void* g) {
  *(v8h*)lds = *(const v8h*)g;
}
__device__ __forceinline__ void async_wait0() {}
#endif

// ---------------------------------------------------------------------------
// gfx1250 Tensor Data Mover: one wave DMAs a whole 2D f16 tile into LDS.
// Pad feature: pad_interval=16 DWORDs (32 halfs = one tile row) +
// pad_amount=4 DWORDs (8 halfs) => padded LDS row stride = 40 halfs (== LDA).
// Descriptor bitfields per CDNA5 ISA ch.8 (D# groups 0..3).
// This toolchain exposes the 6-arg builtin:
//   (uint32x4 g0, int32x8 g1, int32x4 g2, int32x4 g3, int32x8 g4, i32 cpol)
// ---------------------------------------------------------------------------
#if __has_builtin(__builtin_amdgcn_tensor_load_to_lds) && \
    __has_builtin(__builtin_amdgcn_s_wait_tensorcnt)
#define HAVE_TDM 1
__device__ __forceinline__ unsigned lds_addr_of(const void* p) {
  return (unsigned)(unsigned long long)
      (__attribute__((address_space(3))) const void*)p;
}
__device__ __forceinline__ void tdm_load_tile_f16_32w(
    const _Float16* src, unsigned ldsOff, int rows, int strideElems) {
  const unsigned long long ga = (unsigned long long)(const void*)src;
  u4 g0;
  g0[0] = 1u;                                        // count=1 (valid), user
  g0[1] = ldsOff;                                    // lds_addr (bytes)
  g0[2] = (unsigned)(ga & 0xFFFFFFFFu);              // global_addr[31:0]
  g0[3] = (unsigned)((ga >> 32) & 0x1FFFFFFu) | (2u << 30);  // [56:32], type=2
  i8 g1;
  g1[0] = (int)((1u << 16)      // data_size = 1 -> 2 bytes
              | (1u << 20)      // pad_enable
              | (3u << 22)      // pad_interval code 3 = 16 DWORDs (64B row)
              | (3u << 25));    // pad_amount  code 3 = 4 DWORDs (16B pad)
  g1[1] = (int)(((unsigned)strideElems & 0xFFFFu) << 16);   // tensor_dim0 lo
  g1[2] = (int)((((unsigned)strideElems >> 16) & 0xFFFFu)   // tensor_dim0 hi
              | (0x4000u << 16));                            // tensor_dim1 = 16384
  g1[3] = (int)(32u << 16);                                  // tile_dim0 = 32
  g1[4] = (int)((unsigned)rows & 0xFFFFu);                   // tile_dim1 (dim2=0)
  g1[5] = (int)strideElems;                                  // dim0_stride[31:0]
  g1[6] = 0;                                                 // stride hi / dim1_stride
  g1[7] = 0;
  const i4 gz4 = (i4){0, 0, 0, 0};
  const i8 gz8 = (i8){0, 0, 0, 0, 0, 0, 0, 0};
  __builtin_amdgcn_tensor_load_to_lds(g0, g1, gz4, gz4, gz8, 0);
}
__device__ __forceinline__ void tdm_wait0() {
  __builtin_amdgcn_s_wait_tensorcnt(0);
}
#endif

// ---------------------------------------------------------------------------
// WMMA helpers (wave32). 16-bit A/B fragment: lane l holds row/col (l&15);
// half j maps to K = (j<8?0:16) + (l>>4)*8 + (j&7)  => two 16B reads.
// ---------------------------------------------------------------------------
__device__ __forceinline__ v16h frag_ld(const _Float16* rowptr, int hi) {
  v8h a = *(const v8h*)(rowptr + hi * 8);
  v8h b = *(const v8h*)(rowptr + 16 + hi * 8);
  return __builtin_shufflevector(a, b, 0,1,2,3,4,5,6,7,8,9,10,11,12,13,14,15);
}

__device__ __forceinline__ v8f wmma_f16(v16h a, v16h b, v8f c) {
  return __builtin_amdgcn_wmma_f32_16x16x32_f16(false, a, false, b,
                                                (short)0, c, false, false);
}

// ---------------------------------------------------------------------------
// Kernel 0: f32 -> f16 conversion (x, W_qkv, W_o), 4 elems/thread.
// ---------------------------------------------------------------------------
__global__ __launch_bounds__(256) void cvt_f32_f16_kernel(
    const float* __restrict__ s, _Float16* __restrict__ d) {
  const size_t i = ((size_t)blockIdx.x * 256 + threadIdx.x) * 4;
  const v4f x = *(const v4f*)(s + i);
  v4h y;
#pragma unroll
  for (int e = 0; e < 4; ++e) y[e] = (_Float16)x[e];
  *(v4h*)(d + i) = y;
}

// ---------------------------------------------------------------------------
// Unified f16 GEMM: C[m][n] = sum_k A[m][k]*B[n][k].
// Block = 8 waves, tile 128x128x32, wave tile 64x32 (4x2 wmma tiles).
// Double-buffered LDS; A/B slabs staged by the Tensor Data Mover (wave 0
// issues 2 descriptors/step, TENSORcnt), falling back to per-lane async b128.
// MODE 0: scatter f16 into Q/K/V (B,H,L,HD).  MODE 1: f32 to Out.
// ---------------------------------------------------------------------------
#define BM  128
#define BN  128
#define BK  32
#define LDA 40   // 32 + 8 halfs pad (TDM pad reproduces this exactly)

template <int MODE>
__global__ __launch_bounds__(256) void gemm16_kernel(
    const _Float16* __restrict__ Ag, const _Float16* __restrict__ Bg,
    _Float16* __restrict__ Qo, _Float16* __restrict__ Ko,
    _Float16* __restrict__ Vo, float* __restrict__ Out) {
  __shared__ __align__(16) _Float16 As[2][BM * LDA];
  __shared__ __align__(16) _Float16 Bs[2][BN * LDA];
  const int tid  = threadIdx.x;
  const int wid  = tid >> 5, lane = tid & 31;
  const int r    = lane & 15, hi = lane >> 4;
  const int wr   = wid >> 2, wc = wid & 3;
  const int m0   = blockIdx.x * BM;
  const int n0   = blockIdx.y * BN;

  auto stage = [&](int k0, int b) {
#if defined(HAVE_TDM)
    if (wid == 0) {
      tdm_load_tile_f16_32w(Ag + (size_t)m0 * Dc + k0, lds_addr_of(&As[b][0]),
                            BM, Dc);
      tdm_load_tile_f16_32w(Bg + (size_t)n0 * Dc + k0, lds_addr_of(&Bs[b][0]),
                            BN, Dc);
    }
#else
#pragma unroll
    for (int t = 0; t < 2; ++t) {
      const int c   = t * 256 + tid;       // 0..511
      const int row = c >> 2;              // 0..127
      const int col = (c & 3) << 3;        // 0,8,16,24 halfs
      async_cp16(&As[b][row * LDA + col], Ag + (size_t)(m0 + row) * Dc + k0 + col);
      async_cp16(&Bs[b][row * LDA + col], Bg + (size_t)(n0 + row) * Dc + k0 + col);
    }
#endif
  };
  auto stage_wait = [&]() {
#if defined(HAVE_TDM)
    if (wid == 0) tdm_wait0();
#else
    async_wait0();
#endif
  };

  v8f acc[4][2];
#pragma unroll
  for (int i = 0; i < 4; ++i)
#pragma unroll
    for (int j = 0; j < 2; ++j) acc[i][j] = (v8f){0,0,0,0,0,0,0,0};

  stage(0, 0);
  int buf = 0;
  for (int k0 = 0; k0 < Dc; k0 += BK) {
    stage_wait();         // issuing wave's DMA into `buf` complete
    __syncthreads();      // visible to all waves / prev reads done
    if (k0 + BK < Dc) stage(k0 + BK, buf ^ 1);   // prefetch next slab

    v16h af[4], bf[2];
#pragma unroll
    for (int i = 0; i < 4; ++i)
      af[i] = frag_ld(&As[buf][(wr * 64 + i * 16 + r) * LDA], hi);
#pragma unroll
    for (int j = 0; j < 2; ++j)
      bf[j] = frag_ld(&Bs[buf][(wc * 32 + j * 16 + r) * LDA], hi);
#pragma unroll
    for (int i = 0; i < 4; ++i)
#pragma unroll
      for (int j = 0; j < 2; ++j)
        acc[i][j] = wmma_f16(af[i], bf[j], acc[i][j]);
    buf ^= 1;
  }

  if constexpr (MODE == 0) {
    // column n -> {q,k,v}, head h, channel hd; row m -> (b,l)
#pragma unroll
    for (int i = 0; i < 4; ++i) {
#pragma unroll
      for (int j = 0; j < 2; ++j) {
        const int gn    = n0 + wc * 32 + j * 16 + r;
        const int which = gn / Dc;
        const int nn    = gn - which * Dc;
        const int h     = nn / HDc;
        const int hd    = nn - h * HDc;
        _Float16* dst   = which == 0 ? Qo : (which == 1 ? Ko : Vo);
        const int mbase = m0 + wr * 64 + i * 16 + hi * 8;
#pragma unroll
        for (int v = 0; v < 8; ++v) {
          const int m = mbase + v;
          const int b = m >> 10, l = m & (Lc - 1);
          dst[(((size_t)(b * Hc + h)) * Lc + l) * HDc + hd] = (_Float16)acc[i][j][v];
        }
      }
    }
  } else {
#pragma unroll
    for (int i = 0; i < 4; ++i)
#pragma unroll
      for (int j = 0; j < 2; ++j) {
        const int gn    = n0 + wc * 32 + j * 16 + r;
        const int mbase = m0 + wr * 64 + i * 16 + hi * 8;
#pragma unroll
        for (int v = 0; v < 8; ++v)
          Out[(size_t)(mbase + v) * Dc + gn] = acc[i][j][v];
      }
  }
}

// ---------------------------------------------------------------------------
// Kernel 2: SuRoPE on Q (96^-0.5 folded in) and K, in place, f16.
// ---------------------------------------------------------------------------
__global__ __launch_bounds__(256) void rope_kernel(
    _Float16* __restrict__ Qd, _Float16* __restrict__ Kd,
    const float* __restrict__ pos_ids) {
  int idx = blockIdx.x * blockDim.x + threadIdx.x;
  if (idx >= Bc * Hc * Lc * (HDc / 2)) return;
  const int i = idx % 48;
  int t = idx / 48;
  const int l = t & (Lc - 1); t >>= 10;
  const int h = t & (Hc - 1);
  const int b = t >> 5;
  const float pos = pos_ids[b * Lc + l];
  const float inv = __expf(-9.210340371976184f * ((float)(2 * i) / 96.0f));
  const float f   = pos * inv;
  const float s   = __sinf(f) * SCALING_F;
  const float c   = __cosf(f) * SCALING_F;
  const size_t base = ((size_t)((b * Hc + h) * Lc + l)) * HDc;
  const float q0 = (float)Qd[base + i], q1 = (float)Qd[base + i + 48];
  Qd[base + i]      = (_Float16)((q0 * c - q1 * s) * QSCALE_F);
  Qd[base + i + 48] = (_Float16)((q1 * c + q0 * s) * QSCALE_F);
  const float k0 = (float)Kd[base + i], k1 = (float)Kd[base + i + 48];
  Kd[base + i]      = (_Float16)(k0 * c - k1 * s);
  Kd[base + i + 48] = (_Float16)(k1 * c + k0 * s);
}

// ---------------------------------------------------------------------------
// Kernel 3: retention attention.  grid = (B*H, L/128); block = 8 waves.
// Per 32-key step: 6 wmma (QK^T over K=96, 2 col tiles), decay in f32 regs,
// f16 P through per-wave LDS scratch, 6 wmma (P*V). K tile staged with async
// b128 copies into conflict-free 104-half-stride rows (TDM padding cannot
// express 96-half rows); V staged transposed via DS stores.
// ---------------------------------------------------------------------------
#define LDK  104   // 96 + 8 pad
#define LDV  40    // 32 + 8 pad
#define LDSS 40

__global__ __launch_bounds__(256) void attn_kernel(
    const _Float16* __restrict__ Qd, const _Float16* __restrict__ Kd,
    const _Float16* __restrict__ Vd, float* __restrict__ Od) {
  __shared__ __align__(16) _Float16 Ksm[32 * LDK];
  __shared__ __align__(16) _Float16 Vtm[96 * LDV];   // [hd][key]
  __shared__ __align__(16) _Float16 Ssm[8 * 16 * LDSS];
  const int bh = blockIdx.x;              // b*32 + h
  const int h  = bh & (Hc - 1);
  const int b  = bh >> 5;
  const int tid = threadIdx.x, wid = tid >> 5, lane = tid & 31;
  const int r = lane & 15, hi = lane >> 4;
  const int qrow = blockIdx.y * 128 + wid * 16;
  const _Float16* Qb = Qd + (size_t)bh * Lc * HDc;
  const _Float16* Kb = Kd + (size_t)bh * Lc * HDc;
  const _Float16* Vb = Vd + (size_t)bh * Lc * HDc;

  v16h qf[3];
#pragma unroll
  for (int c = 0; c < 3; ++c)
    qf[c] = frag_ld(Qb + (size_t)(qrow + r) * HDc + c * 32, hi);

  // gamma[h] = 1 - exp(linspace(log(1/32), log(1/512), 32))
  const float xs    = -3.4657359027997265f - 0.08943834588069357f * (float)h;
  const float gamma = 1.0f - __expf(xs);
  const float lg    = __logf(gamma);

  v8f acc[6];
#pragma unroll
  for (int t = 0; t < 6; ++t) acc[t] = (v8f){0,0,0,0,0,0,0,0};

  _Float16* Sw = Ssm + wid * 16 * LDSS;

  for (int kb = 0; kb < Lc; kb += 32) {
    __syncthreads();
    // async-stage K rows [kb,kb+32): 384 chunks of 16B
    {
      const int c0  = tid;
      const int r0  = c0 / 12, q0 = (c0 % 12) << 3;
      async_cp16(Ksm + r0 * LDK + q0, Kb + (size_t)(kb + r0) * HDc + q0);
      if (tid < 128) {
        const int c1 = 256 + tid;
        const int r1 = c1 / 12, q1 = (c1 % 12) << 3;
        async_cp16(Ksm + r1 * LDK + q1, Kb + (size_t)(kb + r1) * HDc + q1);
      }
    }
    // stage V transposed: 2 halfs x6 per thread
#pragma unroll
    for (int t = 0; t < 6; ++t) {
      const int flat = tid * 12 + t * 2;
      const int row = flat / 96, col = flat % 96;
      const v2h p = *(const v2h*)(Vb + (size_t)(kb + row) * HDc + col);
      Vtm[col * LDV + row]       = p.x;
      Vtm[(col + 1) * LDV + row] = p.y;
    }
    async_wait0();
    __syncthreads();

    // S = Q * K^T, decay, -> f16 into per-wave scratch
#pragma unroll
    for (int jj = 0; jj < 2; ++jj) {
      v8f s = (v8f){0,0,0,0,0,0,0,0};
#pragma unroll
      for (int c = 0; c < 3; ++c) {
        v16h kf = frag_ld(Ksm + (jj * 16 + r) * LDK + c * 32, hi);
        s = wmma_f16(qf[c], kf, s);
      }
      const int key = kb + jj * 16 + r;   // C layout: lane col = key
#pragma unroll
      for (int v = 0; v < 8; ++v) {
        const int d = (qrow + hi * 8 + v) - key;
        const float dec = (d > 0) ? __expf((float)d * lg) : 1.0f;
        Sw[(hi * 8 + v) * LDSS + jj * 16 + r] = (_Float16)(s[v] * dec);
      }
    }
    // O += P(16x32) * V(32x96); wave-private LDS RAW ordered by DScnt
    v16h pf = frag_ld(Sw + r * LDSS, hi);
#pragma unroll
    for (int t = 0; t < 6; ++t) {
      v16h vf = frag_ld(Vtm + (t * 16 + r) * LDV, hi);
      acc[t] = wmma_f16(pf, vf, acc[t]);
    }
  }

  // o[b*L + q, h*96 + hd] f32
#pragma unroll
  for (int t = 0; t < 6; ++t)
#pragma unroll
    for (int v = 0; v < 8; ++v) {
      const int qg = qrow + hi * 8 + v;
      Od[(size_t)(b * Lc + qg) * Dc + h * HDc + t * 16 + r] = acc[t][v];
    }
}

// ---------------------------------------------------------------------------
// Kernel 4: GroupNorm over 32 consecutive channels; one wave per (row,group).
// ---------------------------------------------------------------------------
__global__ __launch_bounds__(256) void groupnorm_kernel(
    const float* __restrict__ O, _Float16* __restrict__ On) {
  const int gid  = blockIdx.x * 8 + (threadIdx.x >> 5);
  const int lane = threadIdx.x & 31;
  const int g    = gid % 96;
  const int row  = gid / 96;
  const size_t idx = (size_t)row * Dc + g * 32 + lane;
  const float x = O[idx];
  float m = x;
#pragma unroll
  for (int off = 16; off > 0; off >>= 1) m += __shfl_xor(m, off, 32);
  m *= (1.0f / 32.0f);
  const float d = x - m;
  float vv = d * d;
#pragma unroll
  for (int off = 16; off > 0; off >>= 1) vv += __shfl_xor(vv, off, 32);
  vv *= (1.0f / 32.0f);
  On[idx] = (_Float16)(d * __frsqrt_rn(vv + 1e-5f));
}

// ---------------------------------------------------------------------------
extern "C" void kernel_launch(void* const* d_in, const int* in_sizes, int n_in,
                              void* d_out, int out_size, void* d_ws, size_t ws_size,
                              hipStream_t stream) {
  const float* x    = (const float*)d_in[0];   // (B,L,D)
  const float* pos  = (const float*)d_in[1];   // (B,L)
  const float* Wqkv = (const float*)d_in[2];   // (3D, D)
  const float* Wo   = (const float*)d_in[3];   // (D, D)
  float* out = (float*)d_out;                  // (B,L,D)

  const size_t NX   = (size_t)Mrows * Dc;          // 6291456
  const size_t NWQ  = (size_t)3 * Dc * Dc;         // 28311552
  const size_t NWO  = (size_t)Dc * Dc;             // 9437184
  const size_t NQK  = (size_t)Bc * Hc * Lc * HDc;  // 6291456

  char* ws = (char*)d_ws;
  _Float16* X16   = (_Float16*)ws;
  _Float16* WQ16  = X16 + NX;
  _Float16* WO16  = WQ16 + NWQ;
  _Float16* Q     = WO16 + NWO;
  _Float16* K     = Q + NQK;
  _Float16* V     = K + NQK;
  float*    O     = (float*)(V + NQK);
  _Float16* On    = (_Float16*)(O + NQK);

  cvt_f32_f16_kernel<<<NX  / 1024, 256, 0, stream>>>(x,    X16);
  cvt_f32_f16_kernel<<<NWQ / 1024, 256, 0, stream>>>(Wqkv, WQ16);
  cvt_f32_f16_kernel<<<NWO / 1024, 256, 0, stream>>>(Wo,   WO16);

  dim3 g1(Mrows / BM, (3 * Dc) / BN);          // 16 x 72
  gemm16_kernel<0><<<g1, 256, 0, stream>>>(X16, WQ16, Q, K, V, nullptr);

  const int nRope = Bc * Hc * Lc * (HDc / 2);  // 3145728
  rope_kernel<<<nRope / 256, 256, 0, stream>>>(Q, K, pos);

  dim3 g3(Bc * Hc, Lc / 128);                  // 64 x 8
  attn_kernel<<<g3, 256, 0, stream>>>(Q, K, V, O);

  groupnorm_kernel<<<(Mrows * 96) / 8, 256, 0, stream>>>(O, On);

  dim3 g5(Mrows / BM, Dc / BN);                // 16 x 24
  gemm16_kernel<1><<<g5, 256, 0, stream>>>(On, WO16, nullptr, nullptr, nullptr, out);
}
